// KPInv_13443247637186
// MI455X (gfx1250) — matrix operand order repaired and checked by reading, not compiled, not run on hardware
//
#include <hip/hip_runtime.h>
#include <hip/hip_bf16.h>
#include <math.h>

typedef __attribute__((ext_vector_type(16))) __bf16 v16bf;
typedef __attribute__((ext_vector_type(8)))  __bf16 v8bf;
typedef __attribute__((ext_vector_type(8)))  float  v8f;

#define HH 32          // neighbors
#define CC 128         // channels
#define KK 15          // kernel points
#define CPG 16
#define NCOL2 240      // K*CPG
#define NEG_SLOPE 0.1f
#define GN_EPS 1e-5f
#define KSTEPS 4       // 128 / 32

// ---------------------------------------------------------------------------
// One-time prep: f32 -> bf16 row-major (for A operands).
// ---------------------------------------------------------------------------
__global__ void to_bf16(const float* __restrict__ src, __bf16* __restrict__ dst,
                        long n)
{
    long i = (long)blockIdx.x * blockDim.x + threadIdx.x;
    const long stride = (long)gridDim.x * blockDim.x;
    for (; i < n; i += stride) dst[i] = (__bf16)src[i];
}

// ---------------------------------------------------------------------------
// One-time prep: swizzle W (128 x N, f32) into the WMMA B-operand layout:
//   Bsw[((ntile*KSTEPS + ks)*32 + lane)*16 + i] = bf16( W[k][col] )
//   k = ks*32 + (lane>=16 ? 8 : 0) + i + (i>=8 ? 8 : 0),   col = ntile*16 + (lane&15)
// Each lane of the GEMM then loads its 16 B elements as one contiguous 32B.
// ---------------------------------------------------------------------------
__global__ void swizzle_B(const float* __restrict__ W, __bf16* __restrict__ Bsw,
                          int N)
{
    const long total = (long)N * 128;
    long e = (long)blockIdx.x * blockDim.x + threadIdx.x;
    const long stride = (long)gridDim.x * blockDim.x;
    for (; e < total; e += stride) {
        const int i     = (int)(e & 15);
        const int lane  = (int)((e >> 4) & 31);
        const int ks    = (int)((e >> 9) & 3);
        const int ntile = (int)(e >> 11);
        const int k     = ks * 32 + ((lane >> 4) ? 8 : 0) + i + ((i >= 8) ? 8 : 0);
        const int col   = ntile * 16 + (lane & 15);
        Bsw[e] = (__bf16)W[(size_t)k * N + col];
    }
}

// ---------------------------------------------------------------------------
// GEMM: C[M x N] = act(A[M x 128] @ B[128 x N] + bias).
// A: bf16 row-major. Bsw: pre-swizzled bf16 (see swizzle_B).
// One wave per 16x16 tile; K loop fully unrolled -> 4 x v_wmma_f32_16x16x32_bf16.
// A-operand layout (ISA 7.12.2): lane holds row M=lane&15, K-half by lane>=16;
// elems 0..7 -> K=kb..kb+7, elems 8..15 -> K=kb+16..kb+23  (kb = 0 or 8).
// OUT_BF16: write bf16 row-major (feeds next GEMM's A); else write f32.
// ---------------------------------------------------------------------------
template <bool LEAKY, bool OUT_BF16>
__global__ void gemm_wmma(const __bf16* __restrict__ A,
                          const __bf16* __restrict__ Bsw,
                          const float*  __restrict__ bias,
                          float*  __restrict__ Cf,
                          __bf16* __restrict__ Cb,
                          int N)
{
    const int lane = threadIdx.x & 31;
    const int row0 = blockIdx.x * 16;
    const int col0 = blockIdx.y * 16;
    const int hi   = lane >> 4;             // lane half
    const int mrow = row0 + (lane & 15);
    const int ncol = col0 + (lane & 15);
    const int kbA  = hi ? 8 : 0;

    const __bf16* __restrict__ arow = A + (size_t)mrow * 128;
    const v16bf*  __restrict__ bp   =
        (const v16bf*)Bsw + ((size_t)blockIdx.y * KSTEPS) * 32 + lane;

    v8f acc = {};
#pragma unroll
    for (int ks = 0; ks < KSTEPS; ++ks) {
        const int k0 = ks * 32;
        const v8bf alo = *(const v8bf*)(arow + k0 + kbA);
        const v8bf ahi = *(const v8bf*)(arow + k0 + kbA + 16);
        const v16bf a = __builtin_shufflevector(alo, ahi,
            0, 1, 2, 3, 4, 5, 6, 7, 8, 9, 10, 11, 12, 13, 14, 15);
        const v16bf b = bp[ks * 32];
        acc = __builtin_amdgcn_wmma_f32_16x16x32_bf16(
            /*neg_a=*/false, a, /*neg_b=*/false, b,
            /*c_mod=*/(short)0, acc, /*reuse_a=*/false, /*reuse_b=*/false);
    }

    const float bv = bias[ncol];
#pragma unroll
    for (int r = 0; r < 8; ++r) {
        const int m = row0 + r + (hi << 3);
        float v = acc[r] + bv;
        if (LEAKY) v = (v >= 0.0f) ? v : NEG_SLOPE * v;
        if (OUT_BF16) Cb[(size_t)m * N + ncol] = (__bf16)v;
        else          Cf[(size_t)m * N + ncol] = v;
    }
}

// ---------------------------------------------------------------------------
// GroupNorm stats: per kernel-point k, sum / sum-of-squares over (M, CPG).
// stats[0..14]=sum, [16..30]=sumsq (atomic), [32..46]=mu, [48..62]=rsigma.
// ---------------------------------------------------------------------------
__global__ void stats_init(float* __restrict__ stats)
{
    if (threadIdx.x < 64) stats[threadIdx.x] = 0.0f;
}

__global__ void cw_stats(const float* __restrict__ cw,
                         float* __restrict__ stats, int M)
{
    const int t = threadIdx.x;          // 0..239
    const int k = t >> 4;               // group 0..14
    float s = 0.0f, sq = 0.0f;
    for (int m = blockIdx.x; m < M; m += gridDim.x) {
        const float v = cw[(size_t)m * NCOL2 + t];
        s  += v;
        sq += v * v;
    }
    __shared__ float lsum[16], lsq[16];
    if (t < 16) { lsum[t] = 0.0f; lsq[t] = 0.0f; }
    __syncthreads();
    atomicAdd(&lsum[k], s);             // ds_add_f32
    atomicAdd(&lsq[k], sq);
    __syncthreads();
    if (t < KK) {
        atomicAdd(&stats[t], lsum[t]);  // global_atomic_add_f32
        atomicAdd(&stats[16 + t], lsq[t]);
    }
}

__global__ void cw_finalize(float* __restrict__ stats, int M)
{
    const int t = threadIdx.x;
    if (t < KK) {
        const float inv = 1.0f / ((float)M * (float)CPG);
        const float mu  = stats[t] * inv;
        const float var = stats[16 + t] * inv - mu * mu;
        stats[32 + t] = mu;
        stats[48 + t] = rsqrtf(var + GN_EPS);
    }
}

// ---------------------------------------------------------------------------
// Final aggregation: one block per query point (128 threads).
//  Phase A (lanes 0-31): nearest kernel point + linear influence per neighbor
//  Phase B: 512 normalized+influenced weights into LDS
//  Phase C: thread t = channel c: out[c] = sum_h feats[idx[h]][c]*w[h][c>>3]
// s_feats (25.6 MB) is L2-resident on MI455X (192 MB L2), so the 819 MB of
// gather traffic stays on-chip.
// ---------------------------------------------------------------------------
__global__ void kpinv_aggregate(const float* __restrict__ q_pts,
                                const float* __restrict__ s_pts,
                                const float* __restrict__ s_feats,
                                const int*   __restrict__ nidx,
                                const float* __restrict__ k_pts,
                                const float* __restrict__ cw,
                                const float* __restrict__ stats,
                                const float* __restrict__ gn_w,
                                const float* __restrict__ gn_b,
                                float* __restrict__ out,
                                int Npts)
{
    const int m = blockIdx.x;
    const int t = threadIdx.x;          // 0..127

    __shared__ float skp[KK * 3];
    __shared__ int   sidx[HH];
    __shared__ float sinf[HH];
    __shared__ int   skid[HH];
    __shared__ float sw[HH * CPG];      // 512 weights

    if (t < KK * 3) skp[t] = k_pts[t];
    __syncthreads();

    if (t < HH) {
        const int idx = nidx[(size_t)m * HH + t];
        sidx[t] = idx;
        const float qx = q_pts[m * 3 + 0];
        const float qy = q_pts[m * 3 + 1];
        const float qz = q_pts[m * 3 + 2];
        float px, py, pz;
        if (idx < Npts) {
            px = s_pts[(size_t)idx * 3 + 0];
            py = s_pts[(size_t)idx * 3 + 1];
            pz = s_pts[(size_t)idx * 3 + 2];
        } else {                        // padding row = (1e6,1e6,1e6)
            px = py = pz = 1.0e6f;
        }
        const float dx = px - qx, dy = py - qy, dz = pz - qz;
        float best = 1.0e30f;
        int   bk   = 0;
#pragma unroll
        for (int k = 0; k < KK; ++k) {
            const float ex = dx - skp[k * 3 + 0];
            const float ey = dy - skp[k * 3 + 1];
            const float ez = dz - skp[k * 3 + 2];
            const float d  = ex * ex + ey * ey + ez * ez;
            if (d < best) { best = d; bk = k; }
        }
        sinf[t] = fmaxf(1.0f - sqrtf(best), 0.0f);   // SIGMA = 1
        skid[t] = bk;
    }
    __syncthreads();

#pragma unroll
    for (int e = t; e < HH * CPG; e += 128) {
        const int h   = e >> 4;
        const int cpg = e & 15;
        const int k   = skid[h];
        const int col = k * CPG + cpg;
        float v = cw[(size_t)m * NCOL2 + col];
        v = (v - stats[32 + k]) * stats[48 + k] * gn_w[col] + gn_b[col];
        sw[e] = v * sinf[h];
    }
    __syncthreads();

    const int c    = t;
    const int wcol = c >> 3;            // cpg = c / GROUPS
    float acc = 0.0f;
#pragma unroll 4
    for (int h = 0; h < HH; ++h) {
        const int idx = sidx[h];
        if (idx < Npts)                 // padding feature row = zeros
            acc = fmaf(s_feats[(size_t)idx * CC + c], sw[h * CPG + wcol], acc);
    }
    out[(size_t)m * CC + c] = acc;
}

// ---------------------------------------------------------------------------
// Launch. Inputs (setup_inputs order): q_pts, s_pts, s_feats, neighb_inds,
// k_pts, W1, b1, W2, b2, gn_w, gn_b. Output: (M,128) f32.
// Workspace: sfb (N*128 bf16) | hb (M*128 bf16) | w1sw (16K bf16) |
//            w2sw (30K bf16) | cw (M*240 f32) | stats (64 f32)  ~= 74 MB.
// ---------------------------------------------------------------------------
extern "C" void kernel_launch(void* const* d_in, const int* in_sizes, int n_in,
                              void* d_out, int out_size, void* d_ws, size_t ws_size,
                              hipStream_t stream)
{
    const float* q_pts   = (const float*)d_in[0];
    const float* s_pts   = (const float*)d_in[1];
    const float* s_feats = (const float*)d_in[2];
    const int*   nidx    = (const int*)  d_in[3];
    const float* k_pts   = (const float*)d_in[4];
    const float* W1      = (const float*)d_in[5];
    const float* b1      = (const float*)d_in[6];
    const float* W2      = (const float*)d_in[7];
    const float* b2      = (const float*)d_in[8];
    const float* gn_w    = (const float*)d_in[9];
    const float* gn_b    = (const float*)d_in[10];
    float* out = (float*)d_out;

    const int M    = in_sizes[0] / 3;   // 50000 (multiple of 16)
    const int Npts = in_sizes[1] / 3;   // 50000

    // Workspace carve-up (keep每 region 256B-aligned).
    char* p = (char*)d_ws;
    __bf16* sfb  = (__bf16*)p;  p += ((size_t)Npts * CC * 2 + 255) & ~255ULL;
    __bf16* hb   = (__bf16*)p;  p += ((size_t)M * CC * 2 + 255) & ~255ULL;
    __bf16* w1sw = (__bf16*)p;  p += ((size_t)128 * CC * 2 + 255) & ~255ULL;
    __bf16* w2sw = (__bf16*)p;  p += ((size_t)128 * NCOL2 * 2 + 255) & ~255ULL;
    float*  cwb  = (float*)p;   p += ((size_t)M * NCOL2 * 4 + 255) & ~255ULL;
    float*  stats = (float*)p;

    const int mtiles = M / 16;

    stats_init<<<1, 64, 0, stream>>>(stats);

    // One-time bf16 prep (bandwidth-trivial vs. the gather phase).
    to_bf16<<<1024, 256, 0, stream>>>(s_feats, sfb, (long)Npts * CC);
    swizzle_B<<<64, 256, 0, stream>>>(W1, w1sw, CC);
    swizzle_B<<<120, 256, 0, stream>>>(W2, w2sw, NCOL2);

    // h = LeakyReLU(s_feats @ W1 + b1)   (bf16 out, feeds GEMM2's A operand)
    gemm_wmma<true, true><<<dim3(mtiles, CC / 16), 32, 0, stream>>>(
        sfb, w1sw, b1, nullptr, hb, CC);

    // cw = h @ W2 + b2                   (f32 out for GroupNorm stats)
    gemm_wmma<false, false><<<dim3(mtiles, NCOL2 / 16), 32, 0, stream>>>(
        hb, w2sw, b2, cwb, nullptr, NCOL2);

    cw_stats<<<512, NCOL2, 0, stream>>>(cwb, stats, M);
    cw_finalize<<<1, 32, 0, stream>>>(stats, M);

    kpinv_aggregate<<<M, 128, 0, stream>>>(
        q_pts, s_pts, s_feats, nidx, k_pts, cwb, stats, gn_w, gn_b,
        out, Npts);
}
